// StrictCausalAttention_8418135900690
// MI455X (gfx1250) — matrix operand order repaired and checked by reading, not compiled, not run on hardware
//
#include <hip/hip_runtime.h>
#include <hip/hip_bf16.h>
#include <cstdint>
#include <cstddef>

#define D_MODEL 512
#define N_HEADS 8
#define D_HEAD  64
#define NSEQ    2048
#define BATCH   4
#define WINDOW  64
#define NEGV    (-1.0e9f)

typedef _Float16 f16;
typedef __attribute__((ext_vector_type(16))) _Float16 v16h;
typedef __attribute__((ext_vector_type(8)))  float    v8f;

__device__ __forceinline__ int iclamp(int x, int lo, int hi) {
  return x < lo ? lo : (x > hi ? hi : x);
}

__device__ __forceinline__ v8f wmma_f16(v16h a, v16h b, v8f c) {
  // D = A(16x32 f16) * B(32x16 f16) + C(16x16 f32)
  return __builtin_amdgcn_wmma_f32_16x16x32_f16(false, a, false, b, (short)0, c,
                                                false, false);
}

// ---- fragment loaders (wave32, contiguous k = 16*(lane/16)+s; A and B match) -------------
// A 16x32 f16: lanes 0-15 -> row m0+lane, K = {k0..k0+15}
//              lanes 16-31 -> row m0+lane-16, K = {k0+16..k0+31}
__device__ __forceinline__ v16h load_a_f32(const float* __restrict__ src, int ld,
                                           int m0, int k0, int lane) {
  const int row = m0 + (lane & 15);
  const int kb  = (lane >> 4) << 4;                 // 0 or 16
  const float* p = src + (size_t)row * ld + k0 + kb;
  v16h a;
#pragma unroll
  for (int i = 0; i < 8; ++i) { a[i] = (f16)p[i]; a[8 + i] = (f16)p[8 + i]; }
  return a;
}

__device__ __forceinline__ v16h load_a_f16(const f16* __restrict__ src, int ld,
                                           int m0, int k0, int lane) {
  const int row = m0 + (lane & 15);
  const int kb  = (lane >> 4) << 4;
  const f16* p = src + (size_t)row * ld + k0 + kb;
  v16h a;
#pragma unroll
  for (int i = 0; i < 8; ++i) { a[i] = p[i]; a[8 + i] = p[8 + i]; }
  return a;
}

// B 32x16 f16 where B[k][n] = W[n0+n][k0+k] (i.e. x @ W.T): lane's column is
// n0+(lane&15); half-wave selects K block of 16 -> 16 contiguous k-reads.
__device__ __forceinline__ v16h load_b_trans_f32(const float* __restrict__ W, int ld,
                                                 int n0, int k0, int lane) {
  const int n  = n0 + (lane & 15);
  const int kb = (lane >> 4) << 4;                  // 0 or 16
  const float* p = W + (size_t)n * ld + k0 + kb;
  v16h b;
#pragma unroll
  for (int i = 0; i < 16; ++i) b[i] = (f16)p[i];
  return b;
}

// ---- kernel 1: qkv = x @ Wqkv.T + bqkv, scattered to f16 [B][H][N][dh] ---
__global__ void qkv_proj_kernel(const float* __restrict__ x,
                                const float* __restrict__ Wqkv,
                                const float* __restrict__ bqkv,
                                f16* __restrict__ qO, f16* __restrict__ kO,
                                f16* __restrict__ vO) {
  const int lane = threadIdx.x;
  const int m0 = blockIdx.x * 16;                   // row tile in [0, 8192)
  const int n0 = blockIdx.y * 64;                   // col strip in [0, 1536)
  v8f acc[4];
#pragma unroll
  for (int t = 0; t < 4; ++t)
#pragma unroll
    for (int r = 0; r < 8; ++r) acc[t][r] = 0.0f;

  for (int k0 = 0; k0 < D_MODEL; k0 += 32) {
    const v16h a = load_a_f32(x, D_MODEL, m0, k0, lane);
#pragma unroll
    for (int t = 0; t < 4; ++t) {
      const v16h bq = load_b_trans_f32(Wqkv, D_MODEL, n0 + 16 * t, k0, lane);
      acc[t] = wmma_f16(a, bq, acc[t]);
    }
  }

  const int nn = lane & 15, hi = lane >> 4;
#pragma unroll
  for (int t = 0; t < 4; ++t) {
    const int gn   = n0 + 16 * t + nn;              // 0..1535
    const float bias = bqkv[gn];
    const int sel  = gn >> 9;                       // 0=q 1=k 2=v
    const int dcol = gn & 511;
    const int h = dcol >> 6, dd = dcol & 63;
    f16* dst = sel == 0 ? qO : (sel == 1 ? kO : vO);
#pragma unroll
    for (int r = 0; r < 8; ++r) {
      const int m   = m0 + r + 8 * hi;
      const int bb  = m >> 11;                      // batch
      const int seq = m & 2047;
      dst[(((size_t)(bb * N_HEADS + h)) * NSEQ + seq) * D_HEAD + dd] =
          (f16)(acc[t][r] + bias);
    }
  }
}

// ---- kernel 2: windowed attention, one wave per (b, h, 16-query block) ---
__global__ void attn_kernel(const f16* __restrict__ qF, const f16* __restrict__ kF,
                            const f16* __restrict__ vF,
                            const unsigned char* __restrict__ pad,
                            f16* __restrict__ aOut) {
  __shared__ float S[16][80];                       // fp32 scores, 80-key band
  __shared__ f16   P[16][96];                       // f16 probs, padded to 96
  const int lane = threadIdx.x;
  int idx = blockIdx.x;
  const int qb = idx & 127;  idx >>= 7;
  const int h  = idx & 7;    idx >>= 3;
  const int b  = idx;
  const size_t headOff = ((size_t)(b * N_HEADS + h)) * NSEQ * D_HEAD;
  const f16* qH = qF + headOff;
  const f16* kH = kF + headOff;
  const f16* vH = vF + headOff;
  const int r0  = qb * 16;
  const int kb0 = r0 - WINDOW;                      // band start (may be < 0)
  const int nn = lane & 15, hi = lane >> 4;

  const v16h aq0 = load_a_f16(qH, D_HEAD, r0, 0,  lane);
  const v16h aq1 = load_a_f16(qH, D_HEAD, r0, 32, lane);

  // S = (Q K^T) * dh^-1/2 over 5 key tiles
#pragma unroll
  for (int j = 0; j < 5; ++j) {
    const int kblk = kb0 + 16 * j;
    v8f s;
#pragma unroll
    for (int r = 0; r < 8; ++r) s[r] = 0.0f;
    if (kblk + 15 >= 0) {                           // wave-uniform guard
      // B[k=d][n=key] = K[key][d]; keys clamped, masked later in softmax
      const int key = iclamp(kblk + nn, 0, NSEQ - 1);
      const int db  = hi << 4;
      const f16* p0 = kH + (size_t)key * D_HEAD + db;
      v16h b0, b1;
#pragma unroll
      for (int i = 0; i < 16; ++i) { b0[i] = p0[i]; b1[i] = p0[32 + i]; }
      s = wmma_f16(aq0, b0, s);
      s = wmma_f16(aq1, b1, s);
    }
#pragma unroll
    for (int r = 0; r < 8; ++r) S[r + 8 * hi][16 * j + nn] = s[r] * 0.125f;
  }
  __syncthreads();

  // fp32 softmax, one row per lane (lanes 0-15)
  if (lane < 16) {
    const int row = r0 + lane;
    float mx = -3.0e38f;
    for (int c = 0; c < 80; ++c) {
      const int key = kb0 + c;
      bool ok = (key >= 0) && (key <= row) && (row - key <= WINDOW);
      if (ok) ok = pad[(size_t)b * NSEQ + key] != 0;
      const float sv = ok ? S[lane][c] : NEGV;
      S[lane][c] = sv;
      mx = fmaxf(mx, sv);
    }
    float sum = 0.0f;
    for (int c = 0; c < 80; ++c) {
      const float e = __expf(S[lane][c] - mx);
      S[lane][c] = e;
      sum += e;
    }
    const float inv = 1.0f / sum;
    for (int c = 0; c < 80; ++c)  P[lane][c] = (f16)(S[lane][c] * inv);
    for (int c = 80; c < 96; ++c) P[lane][c] = (f16)0.0f;
  }
  __syncthreads();

  // O = P(16x96) @ V(96x64); zero-padded P kills clamped/out-of-band keys
  v8f o[4];
#pragma unroll
  for (int t = 0; t < 4; ++t)
#pragma unroll
    for (int r = 0; r < 8; ++r) o[t][r] = 0.0f;

#pragma unroll
  for (int stp = 0; stp < 3; ++stp) {
    const int kO = stp * 32;
    v16h ap;                                        // A fragment from LDS
    {
      const int kb = hi << 4;
#pragma unroll
      for (int i = 0; i < 8; ++i) {
        ap[i]     = P[nn][kO + kb + i];
        ap[8 + i] = P[nn][kO + kb + 8 + i];
      }
    }
#pragma unroll
    for (int t = 0; t < 4; ++t) {
      // B[k=key_rel][n=d] = V[key][d]
      v16h bv;
      const int d    = 16 * t + nn;
      const int keyb = kb0 + kO + (hi << 4);
#pragma unroll
      for (int i = 0; i < 16; ++i) {
        const int key = iclamp(keyb + i, 0, NSEQ - 1);
        bv[i] = vH[(size_t)key * D_HEAD + d];
      }
      o[t] = wmma_f16(ap, bv, o[t]);
    }
  }

  // store heads re-interleaved: [b][seq][h*dh + d]  (f16)
#pragma unroll
  for (int t = 0; t < 4; ++t)
#pragma unroll
    for (int r = 0; r < 8; ++r) {
      const int seq = r0 + r + 8 * hi;
      aOut[((size_t)b * NSEQ + seq) * D_MODEL + h * D_HEAD + 16 * t + nn] =
          (f16)o[t][r];
    }
}

// ---- kernel 3: out = att @ Wout.T + bout  (fp32 output) ------------------
__global__ void out_proj_kernel(const f16* __restrict__ att,
                                const float* __restrict__ Wout,
                                const float* __restrict__ bout,
                                float* __restrict__ out) {
  const int lane = threadIdx.x;
  const int m0 = blockIdx.x * 16;
  const int n0 = blockIdx.y * 64;
  v8f acc[4];
#pragma unroll
  for (int t = 0; t < 4; ++t)
#pragma unroll
    for (int r = 0; r < 8; ++r) acc[t][r] = 0.0f;

  for (int k0 = 0; k0 < D_MODEL; k0 += 32) {
    const v16h a = load_a_f16(att, D_MODEL, m0, k0, lane);
#pragma unroll
    for (int t = 0; t < 4; ++t) {
      const v16h b = load_b_trans_f32(Wout, D_MODEL, n0 + 16 * t, k0, lane);
      acc[t] = wmma_f16(a, b, acc[t]);
    }
  }

  const int nn = lane & 15, hi = lane >> 4;
#pragma unroll
  for (int t = 0; t < 4; ++t) {
    const int gn = n0 + 16 * t + nn;
    const float bias = bout[gn];
#pragma unroll
    for (int r = 0; r < 8; ++r) {
      const int m = m0 + r + 8 * hi;
      out[(size_t)m * D_MODEL + gn] = acc[t][r] + bias;
    }
  }
}

extern "C" void kernel_launch(void* const* d_in, const int* in_sizes, int n_in,
                              void* d_out, int out_size, void* d_ws, size_t ws_size,
                              hipStream_t stream) {
  const float* x            = (const float*)d_in[0];
  const unsigned char* pad  = (const unsigned char*)d_in[1];  // jnp bool = 1B
  const float* Wqkv         = (const float*)d_in[2];
  const float* bqkv         = (const float*)d_in[3];
  const float* Wout         = (const float*)d_in[4];
  const float* bout         = (const float*)d_in[5];
  float* out                = (float*)d_out;

  // workspace: q,k,v [B][H][N][dh] f16 + att [B*N][D] f16 = 32 MB total
  const size_t elems = (size_t)BATCH * N_HEADS * NSEQ * D_HEAD;  // 4,194,304
  f16* qF = (f16*)d_ws;
  f16* kF = qF + elems;
  f16* vF = kF + elems;
  f16* aF = vF + elems;

  qkv_proj_kernel<<<dim3(512, 24), 32, 0, stream>>>(x, Wqkv, bqkv, qF, kF, vF);
  attn_kernel<<<dim3(BATCH * N_HEADS * (NSEQ / 16)), 32, 0, stream>>>(qF, kF, vF, pad, aF);
  out_proj_kernel<<<dim3(512, 8), 32, 0, stream>>>(aF, Wout, bout, out);
}
